// BloomAttention_48722109006296
// MI455X (gfx1250) — matrix operand order repaired and checked by reading, not compiled, not run on hardware
//
#include <hip/hip_runtime.h>

// ---------------- problem constants ----------------
#define NUM_HEADS 16
#define HEAD_DIM  64
#define HIDDEN    1024
#define BATCH     2
#define SEQ       2048
#define INV_NORM  0.125f
#define TPAD      40        // padded LDS row stride (elements); 80B, 16B-aligned

typedef __attribute__((ext_vector_type(16))) __bf16 v16bf;
typedef __attribute__((ext_vector_type(8)))  __bf16 v8bf;
typedef __attribute__((ext_vector_type(4)))  __bf16 v4bf;
typedef __attribute__((ext_vector_type(8)))  float  v8f;

// ---------------- helpers ----------------
static __device__ __forceinline__ v8f wmma_bf16(v16bf a, v16bf b, v8f c) {
  // D = A(16x32 bf16) * B(32x16 bf16) + C(16x16 f32)
  return __builtin_amdgcn_wmma_f32_16x16x32_bf16(
      /*neg_a=*/false, a, /*neg_b=*/false, b,
      /*c_mod=*/(short)0, c, /*reuse_a=*/false, /*reuse_b=*/false);
}

static __device__ __forceinline__ v16bf concat8(v8bf a, v8bf b) {
  return __builtin_shufflevector(a, b, 0, 1, 2, 3, 4, 5, 6, 7,
                                       8, 9, 10, 11, 12, 13, 14, 15);
}
static __device__ __forceinline__ v16bf load_bf16_16(const __bf16* p) {
  return concat8(*(const v8bf*)(p), *(const v8bf*)(p + 8));
}
static __device__ __forceinline__ v16bf load_bf16_2x8(const __bf16* p0, const __bf16* p1) {
  return concat8(*(const v8bf*)(p0), *(const v8bf*)(p1));
}
// hw-convert split: f = hi + lo (bf16x2 error compensation)
struct bfpair { __bf16 h, l; };
static __device__ __forceinline__ bfpair split1(float f) {
  bfpair r;
  r.h = (__bf16)f;
  r.l = (__bf16)(f - (float)r.h);
  return r;
}

// Async copy one 16B chunk from global (base + 32-bit offset, GVS mode) to LDS.
// Tracked by ASYNCcnt (CDNA5 ISA).
static __device__ __forceinline__ void async_g2l_b128(const __bf16* lds_dst,
                                                      const __bf16* gbase,
                                                      unsigned gbyteoff) {
  // generic LDS pointer: low 32 bits are the LDS byte address
  unsigned ldsoff = (unsigned)(size_t)lds_dst;
  unsigned long long base = (unsigned long long)gbase;
  asm volatile("global_load_async_to_lds_b128 %0, %1, %2"
               :: "v"(ldsoff), "v"(gbyteoff), "s"(base) : "memory");
}
static __device__ __forceinline__ void wait_asynccnt_2() {
  asm volatile("s_wait_asynccnt 2" ::: "memory");
}
static __device__ __forceinline__ void wait_asynccnt_0() {
  asm volatile("s_wait_asynccnt 0" ::: "memory");
}

// =====================================================================
// Kernel 0: elementwise fp32 -> bf16 hi/lo split (run once per tensor)
// =====================================================================
__global__ __launch_bounds__(256) void split_bf16_kernel(
    const float* __restrict__ src, __bf16* __restrict__ hi,
    __bf16* __restrict__ lo, int n4) {
  const int i = blockIdx.x * 256 + threadIdx.x;
  if (i >= n4) return;
  float4 v = ((const float4*)src)[i];
  float t[4] = {v.x, v.y, v.z, v.w};
  v4bf h, l;
#pragma unroll
  for (int j = 0; j < 4; ++j) {
    bfpair p = split1(t[j]);
    h[j] = p.h;
    l[j] = p.l;
  }
  ((v4bf*)hi)[i] = h;
  ((v4bf*)lo)[i] = l;
}

// =====================================================================
// Kernel 1: fused QKV GEMM.  fused = x @ Wqkv^T + b (M=4096,N=3072,K=1024).
// Block = 8 waves sharing one 64-col N chunk (one head,part); waves split M.
// W tile (64x32, hi+lo) staged in LDS via async copies, double buffered.
// Q,K row-major [bh][s][d]; V transposed [bh][d][s]; all bf16 hi/lo.
// =====================================================================
__global__ __launch_bounds__(256) void qkv_wmma_kernel(
    const __bf16* __restrict__ Xh, const __bf16* __restrict__ Xl,
    const __bf16* __restrict__ Wh, const __bf16* __restrict__ Wl,
    const float* __restrict__ bias,
    __bf16* __restrict__ Qh, __bf16* __restrict__ Ql,
    __bf16* __restrict__ Kh, __bf16* __restrict__ Kl,
    __bf16* __restrict__ Vth, __bf16* __restrict__ Vtl) {
  __shared__ __bf16 smem[2][2][64 * TPAD];                // 20 KB

  const int tid  = threadIdx.x;
  const int lane = tid & 31;
  const int lm   = lane & 15;
  const int half = lane >> 4;
  const int w    = tid >> 5;                              // wave in block
  const int mtg  = blockIdx.x / 48;                       // M group (8 tiles)
  const int nc   = blockIdx.x % 48;                       // shared 64-col chunk
  const int mt   = mtg * 8 + w;                           // this wave's M tile
  const int head = nc / 3, part = nc % 3;

  // async stage mapping: thread -> (row 0..63, 16B chunk 0..3)
  const int srow = tid >> 2;
  const int sch  = tid & 3;
  const unsigned gW = (unsigned)(((nc * 64 + srow) * HIDDEN + sch * 8) * 2);
  const int soff = srow * TPAD + sch * 8;

  const size_t aRow = (size_t)(mt * 16 + lm) * HIDDEN;    // A-operand row

  v8f acc[4];
#pragma unroll
  for (int i = 0; i < 4; ++i)
#pragma unroll
    for (int r = 0; r < 8; ++r) acc[i][r] = 0.0f;

  // prologue: stage kb=0 into buffer 0
  async_g2l_b128(&smem[0][0][soff], Wh, gW);
  async_g2l_b128(&smem[0][1][soff], Wl, gW);

  int cur = 0;
  for (int kb = 0; kb < HIDDEN; kb += 32) {
    const int kbn = kb + 32;
    if (kbn < HIDDEN) {                       // stage next tile (overlapped)
      async_g2l_b128(&smem[cur ^ 1][0][soff], Wh, gW + (unsigned)(kbn * 2));
      async_g2l_b128(&smem[cur ^ 1][1][soff], Wl, gW + (unsigned)(kbn * 2));
      wait_asynccnt_2();                      // oldest 2 (current buf) done
    } else {
      wait_asynccnt_0();
    }
    __syncthreads();                          // all waves' stages visible

    const size_t ao = aRow + kb + half * 8;
    v16bf ah = load_bf16_2x8(Xh + ao, Xh + ao + 16);
    v16bf al = load_bf16_2x8(Xl + ao, Xl + ao + 16);
    __builtin_prefetch(Xh + ao + 128, 0, 0);

    const __bf16* sh = &smem[cur][0][0];
    const __bf16* sl = &smem[cur][1][0];
#pragma unroll
    for (int nt = 0; nt < 4; ++nt) {
      const int bo = (nt * 16 + lm) * TPAD + half * 8;
      v16bf bh = load_bf16_2x8(sh + bo, sh + bo + 16);
      v16bf bl = load_bf16_2x8(sl + bo, sl + bo + 16);
      acc[nt] = wmma_bf16(ah, bh, acc[nt]);
      acc[nt] = wmma_bf16(ah, bl, acc[nt]);
      acc[nt] = wmma_bf16(al, bh, acc[nt]);
    }
    __syncthreads();                          // buffer reusable
    cur ^= 1;
  }

  const int tok0 = mt * 16 + half * 8;        // first of this lane's 8 C rows
  const int b    = tok0 >> 11;
  const int s0   = tok0 & (SEQ - 1);
  const int bh_i = b * NUM_HEADS + head;

#pragma unroll
  for (int nt = 0; nt < 4; ++nt) {
    const int n  = nc * 64 + nt * 16 + lm;
    const int d  = nt * 16 + lm;              // 0..63 within head
    const float bv = bias[n];
    if (part == 2) {                          // V: store transposed [bh][d][s]
      v8bf ph, pl;
#pragma unroll
      for (int r = 0; r < 8; ++r) {
        bfpair p = split1(acc[nt][r] + bv);
        ph[r] = p.h;
        pl[r] = p.l;
      }
      size_t idx = ((size_t)bh_i * HEAD_DIM + d) * SEQ + s0;
      *(v8bf*)(Vth + idx) = ph;
      *(v8bf*)(Vtl + idx) = pl;
    } else {                                  // Q / K: row-major [bh][s][d]
      __bf16* Dh = (part == 0) ? Qh : Kh;
      __bf16* Dl = (part == 0) ? Ql : Kl;
#pragma unroll
      for (int r = 0; r < 8; ++r) {
        bfpair p = split1(acc[nt][r] + bv);
        size_t idx = ((size_t)bh_i * SEQ + (s0 + r)) * HEAD_DIM + d;
        Dh[idx] = p.h;
        Dl[idx] = p.l;
      }
    }
  }
}

// =====================================================================
// Kernel 2: causal flash attention with ALiBi, per (bh, 16-query tile) wave.
// scores^T tiles (key x query) feed probs@V directly as A-operand.
// Writes ctx as bf16 hi/lo so the dense GEMM needs no conversion.
// =====================================================================
__global__ __launch_bounds__(256) void attn_wmma_kernel(
    const __bf16* __restrict__ Qh, const __bf16* __restrict__ Ql,
    const __bf16* __restrict__ Kh, const __bf16* __restrict__ Kl,
    const __bf16* __restrict__ Vth, const __bf16* __restrict__ Vtl,
    const float* __restrict__ alibi,
    __bf16* __restrict__ Ch, __bf16* __restrict__ Cl) {
  const int lane = threadIdx.x & 31;
  const int lm   = lane & 15;
  const int half = lane >> 4;
  const int wid  = blockIdx.x * 8 + (threadIdx.x >> 5);   // 4096 waves
  const int bh   = wid >> 7;                              // 0..31
  const int qt   = wid & 127;
  const int qbase = qt * 16;
  const int b    = bh >> 4;

  const size_t qkBase = (size_t)bh * SEQ * HEAD_DIM;      // [bh][s][d]
  const size_t vBase  = (size_t)bh * HEAD_DIM * SEQ;      // [bh][d][s]
  const float* al     = alibi + (size_t)bh * SEQ;

  // Q as B-operand (d x query), two 32-wide d tiles, hi/lo
  v16bf qbh[2], qbl[2];
#pragma unroll
  for (int dt = 0; dt < 2; ++dt) {
    size_t off = qkBase + (size_t)(qbase + lm) * HEAD_DIM + dt * 32 + half * 8;
    qbh[dt] = load_bf16_2x8(Qh + off, Qh + off + 16);
    qbl[dt] = load_bf16_2x8(Ql + off, Ql + off + 16);
  }

  float m = -1e30f, lsumAcc = 0.0f;
  v8f cacc[4];
#pragma unroll
  for (int i = 0; i < 4; ++i)
#pragma unroll
    for (int r = 0; r < 8; ++r) cacc[i][r] = 0.0f;

  const int kend = qbase + 16;                 // causal: keys < qbase+16
  for (int kb = 0; kb < kend; kb += 32) {
    // ---- scores^T: two 16-key tiles ----
    v8f sacc[2];
#pragma unroll
    for (int t = 0; t < 2; ++t) {
#pragma unroll
      for (int r = 0; r < 8; ++r) sacc[t][r] = 0.0f;
      const int krow = kb + t * 16 + lm;       // A row = key
#pragma unroll
      for (int dt = 0; dt < 2; ++dt) {
        size_t off = qkBase + (size_t)krow * HEAD_DIM + dt * 32 + half * 8;
        v16bf kah = load_bf16_2x8(Kh + off, Kh + off + 16);
        v16bf kal = load_bf16_2x8(Kl + off, Kl + off + 16);
        sacc[t] = wmma_bf16(kah, qbh[dt], sacc[t]);
        sacc[t] = wmma_bf16(kah, qbl[dt], sacc[t]);
        sacc[t] = wmma_bf16(kal, qbh[dt], sacc[t]);
      }
    }

    // ---- scale + alibi + causal mask + online softmax ----
    const int qidx = qbase + lm;               // this lane's query
    float p[2][8];
    float tmax = -1e30f;
#pragma unroll
    for (int t = 0; t < 2; ++t) {
      const int k0 = kb + t * 16 + half * 8;   // first key of this lane's rows
      float4 a0 = *(const float4*)(al + k0);
      float4 a1 = *(const float4*)(al + k0 + 4);
      float av[8] = {a0.x, a0.y, a0.z, a0.w, a1.x, a1.y, a1.z, a1.w};
#pragma unroll
      for (int r = 0; r < 8; ++r) {
        const int key = k0 + r;
        float s = sacc[t][r] * INV_NORM + av[r];
        s = (key <= qidx) ? s : -1e30f;
        p[t][r] = s;
        tmax = fmaxf(tmax, s);
      }
    }
    tmax = fmaxf(tmax, __shfl_xor(tmax, 16, 32));
    const float mnew = fmaxf(m, tmax);
    const float corr = __expf(m - mnew);
    float ls = 0.0f;
#pragma unroll
    for (int t = 0; t < 2; ++t)
#pragma unroll
      for (int r = 0; r < 8; ++r) { p[t][r] = __expf(p[t][r] - mnew); ls += p[t][r]; }
    ls += __shfl_xor(ls, 16, 32);
    lsumAcc = lsumAcc * corr + ls;
    m = mnew;

    // rescale ctx accumulator rows (row q = r + 8*half)
#pragma unroll
    for (int r = 0; r < 8; ++r) {
      float rc = __shfl(corr, r + 8 * half, 32);
#pragma unroll
      for (int nt = 0; nt < 4; ++nt) cacc[nt][r] *= rc;
    }

    // probs -> bf16 A-operand (t0 -> elems 0..7, t1 -> elems 8..15)
    v16bf pah, pal;
#pragma unroll
    for (int r = 0; r < 8; ++r) {
      bfpair p0 = split1(p[0][r]);
      bfpair p1 = split1(p[1][r]);
      pah[r] = p0.h; pah[r + 8] = p1.h;
      pal[r] = p0.l; pal[r + 8] = p1.l;
    }

    // ---- ctx += probs @ V  (V^T rows give contiguous B-operand loads) ----
#pragma unroll
    for (int nt = 0; nt < 4; ++nt) {
      size_t off = vBase + (size_t)(nt * 16 + lm) * SEQ + kb + half * 8;
      v16bf vbh = load_bf16_2x8(Vth + off, Vth + off + 16);
      v16bf vbl = load_bf16_2x8(Vtl + off, Vtl + off + 16);
      cacc[nt] = wmma_bf16(pah, vbh, cacc[nt]);
      cacc[nt] = wmma_bf16(pah, vbl, cacc[nt]);
      cacc[nt] = wmma_bf16(pal, vbh, cacc[nt]);
    }
  }

  // ---- normalize, merge heads, store bf16 hi/lo ctx [B*S][HIDDEN] ----
  const float linv = 1.0f / lsumAcc;
  const int h = bh & 15;
#pragma unroll
  for (int r = 0; r < 8; ++r) {
    float rl = __shfl(linv, r + 8 * half, 32);
    const int tok = b * SEQ + qbase + r + 8 * half;
#pragma unroll
    for (int nt = 0; nt < 4; ++nt) {
      const int col = h * HEAD_DIM + nt * 16 + lm;
      bfpair p = split1(cacc[nt][r] * rl);
      Ch[(size_t)tok * HIDDEN + col] = p.h;
      Cl[(size_t)tok * HIDDEN + col] = p.l;
    }
  }
}

// =====================================================================
// Kernel 3: dense GEMM: out = ctx @ Wd^T + bd + residual (4096x1024x1024)
// Same block-cooperative async-LDS staging as the QKV GEMM.
// =====================================================================
__global__ __launch_bounds__(256) void dense_wmma_kernel(
    const __bf16* __restrict__ Ch, const __bf16* __restrict__ Cl,
    const __bf16* __restrict__ Wh, const __bf16* __restrict__ Wl,
    const float* __restrict__ bias, const float* __restrict__ residual,
    float* __restrict__ out) {
  __shared__ __bf16 smem[2][2][64 * TPAD];                // 20 KB

  const int tid  = threadIdx.x;
  const int lane = tid & 31;
  const int lm   = lane & 15;
  const int half = lane >> 4;
  const int w    = tid >> 5;
  const int mtg  = blockIdx.x / 16;
  const int nc   = blockIdx.x % 16;                       // shared 64-col chunk
  const int mt   = mtg * 8 + w;

  const int srow = tid >> 2;
  const int sch  = tid & 3;
  const unsigned gW = (unsigned)(((nc * 64 + srow) * HIDDEN + sch * 8) * 2);
  const int soff = srow * TPAD + sch * 8;

  const size_t aRow = (size_t)(mt * 16 + lm) * HIDDEN;

  v8f acc[4];
#pragma unroll
  for (int i = 0; i < 4; ++i)
#pragma unroll
    for (int r = 0; r < 8; ++r) acc[i][r] = 0.0f;

  async_g2l_b128(&smem[0][0][soff], Wh, gW);
  async_g2l_b128(&smem[0][1][soff], Wl, gW);

  int cur = 0;
  for (int kb = 0; kb < HIDDEN; kb += 32) {
    const int kbn = kb + 32;
    if (kbn < HIDDEN) {
      async_g2l_b128(&smem[cur ^ 1][0][soff], Wh, gW + (unsigned)(kbn * 2));
      async_g2l_b128(&smem[cur ^ 1][1][soff], Wl, gW + (unsigned)(kbn * 2));
      wait_asynccnt_2();
    } else {
      wait_asynccnt_0();
    }
    __syncthreads();

    const size_t ao = aRow + kb + half * 8;
    v16bf ah = load_bf16_2x8(Ch + ao, Ch + ao + 16);
    v16bf al = load_bf16_2x8(Cl + ao, Cl + ao + 16);
    __builtin_prefetch(Ch + ao + 128, 0, 0);

    const __bf16* sh = &smem[cur][0][0];
    const __bf16* sl = &smem[cur][1][0];
#pragma unroll
    for (int nt = 0; nt < 4; ++nt) {
      const int bo = (nt * 16 + lm) * TPAD + half * 8;
      v16bf bh = load_bf16_2x8(sh + bo, sh + bo + 16);
      v16bf bl = load_bf16_2x8(sl + bo, sl + bo + 16);
      acc[nt] = wmma_bf16(ah, bh, acc[nt]);
      acc[nt] = wmma_bf16(ah, bl, acc[nt]);
      acc[nt] = wmma_bf16(al, bh, acc[nt]);
    }
    __syncthreads();
    cur ^= 1;
  }

  const int tok0 = mt * 16 + half * 8;
#pragma unroll
  for (int nt = 0; nt < 4; ++nt) {
    const int n = nc * 64 + nt * 16 + lm;
    const float bv = bias[n];
#pragma unroll
    for (int r = 0; r < 8; ++r) {
      size_t idx = (size_t)(tok0 + r) * HIDDEN + n;
      out[idx] = acc[nt][r] + bv + residual[idx];
    }
  }
}

// =====================================================================
extern "C" void kernel_launch(void* const* d_in, const int* in_sizes, int n_in,
                              void* d_out, int out_size, void* d_ws, size_t ws_size,
                              hipStream_t stream) {
  (void)in_sizes; (void)n_in; (void)out_size; (void)ws_size;
  const float* X     = (const float*)d_in[0];   // hidden_states (1,B,S,H)
  const float* resid = (const float*)d_in[1];   // residual (B,S,H)
  const float* alibi = (const float*)d_in[2];   // (B*H,1,S)
  // d_in[3] attention_mask: causal, computed analytically in-kernel
  const float* Wqkv  = (const float*)d_in[4];
  const float* bqkv  = (const float*)d_in[5];
  const float* Wd    = (const float*)d_in[6];
  const float* bd    = (const float*)d_in[7];

  char* ws = (char*)d_ws;
  const size_t MB = 1024 * 1024;
  // bf16 hi/lo scratch tensors
  __bf16* Qh  = (__bf16*)(ws + 0 * MB);
  __bf16* Ql  = (__bf16*)(ws + 8 * MB);
  __bf16* Kh  = (__bf16*)(ws + 16 * MB);
  __bf16* Kl  = (__bf16*)(ws + 24 * MB);
  __bf16* Vth = (__bf16*)(ws + 32 * MB);
  __bf16* Vtl = (__bf16*)(ws + 40 * MB);
  __bf16* Xh  = (__bf16*)(ws + 48 * MB);
  __bf16* Xl  = (__bf16*)(ws + 56 * MB);
  __bf16* Wqh = (__bf16*)(ws + 64 * MB);
  __bf16* Wql = (__bf16*)(ws + 70 * MB);
  __bf16* Wdh = (__bf16*)(ws + 76 * MB);
  __bf16* Wdl = (__bf16*)(ws + 78 * MB);
  __bf16* Ctxh= (__bf16*)(ws + 80 * MB);
  __bf16* Ctxl= (__bf16*)(ws + 88 * MB);     // ends at 96 MB

  dim3 blk(256);
  const int nX  = BATCH * SEQ * HIDDEN / 4;       // 1,048,576 float4s
  const int nWq = 3 * HIDDEN * HIDDEN / 4;        //   786,432
  const int nWd = HIDDEN * HIDDEN / 4;            //   262,144

  split_bf16_kernel<<<(nX  + 255) / 256, blk, 0, stream>>>(X, Xh, Xl, nX);
  split_bf16_kernel<<<(nWq + 255) / 256, blk, 0, stream>>>(Wqkv, Wqh, Wql, nWq);
  split_bf16_kernel<<<(nWd + 255) / 256, blk, 0, stream>>>(Wd, Wdh, Wdl, nWd);

  qkv_wmma_kernel<<<1536, blk, 0, stream>>>(Xh, Xl, Wqh, Wql, bqkv,
                                            Qh, Ql, Kh, Kl, Vth, Vtl);
  attn_wmma_kernel<<<512, blk, 0, stream>>>(Qh, Ql, Kh, Kl, Vth, Vtl, alibi,
                                            Ctxh, Ctxl);
  dense_wmma_kernel<<<512, blk, 0, stream>>>(Ctxh, Ctxl, Wdh, Wdl, bd, resid,
                                             (float*)d_out);
}
